// StreamingmLSTMBlock_54485955117302
// MI455X (gfx1250) — matrix operand order repaired and checked by reading, not compiled, not run on hardware
//
#include <hip/hip_runtime.h>
#include <hip/hip_bf16.h>
#include <math.h>

// ---------------------------------------------------------------------------
// StreamingmLSTMBlock for MI455X (gfx1250, wave32, WMMA).
// Persistent-kernel scan; bf16 WMMA GEMMs with f32 accumulate.
// Weights are converted once per launch to bf16 AND pre-swizzled into the
// CDNA5 WMMA B-fragment layout: per (n-tile, k-block) a 1KB block ordered
// [lane][16 elems], so each lane's fragment is a 32B contiguous chunk ->
// two global_load_b128 per fragment, fully coalesced per wave.
// State (C,n,m) lives in d_out and is updated in place each step.
// ---------------------------------------------------------------------------

#define B_    8
#define S_    1024
#define INP   768
#define HID   768
#define H_    8
#define D_    96
#define UP    1536
#define FOUT  3088
#define CAPG  15.0f
#define EPSLN 1e-6f

#define NWG   64
#define NTH   256

typedef __attribute__((ext_vector_type(16))) __bf16 v16bf;
typedef __attribute__((ext_vector_type(8)))  float  v8f;

union FragU { unsigned short u[16]; uint4 q[2]; v16bf v; };

__device__ __forceinline__ unsigned short f2bf(float f) {
  unsigned u = __float_as_uint(f);
  unsigned r = u + 0x7FFFu + ((u >> 16) & 1u);   // round-to-nearest-even
  return (unsigned short)(r >> 16);
}

// ---------------- workspace layout (bytes) ----------------
constexpr size_t OFF_SYNC  = 0;                                    // 2 ints
constexpr size_t OFF_XNB   = 256;                                  // u16 [16][INP]
constexpr size_t OFF_XT    = OFF_XNB   + (size_t)16 * INP  * 2;    // f32 [8][UP]
constexpr size_t OFF_RT    = OFF_XT    + (size_t)B_ * UP   * 4;    // f32 [8][HID]
constexpr size_t OFF_XCB   = OFF_RT    + (size_t)B_ * HID  * 4;    // u16 [16][UP]
constexpr size_t OFF_FUSED = OFF_XCB   + (size_t)16 * UP   * 2;    // f32 [8][FOUT]
constexpr size_t OFF_XSKIP = OFF_FUSED + (size_t)B_ * FOUT * 4;    // f32 [8][HID]
constexpr size_t OFF_HBUF  = OFF_XSKIP + (size_t)B_ * HID  * 4;    // f32 [8][HID]
constexpr size_t OFF_HNB   = OFF_HBUF  + (size_t)B_ * HID  * 4;    // u16 [16][HID]
constexpr size_t OFF_WL    = OFF_HNB   + (size_t)16 * HID  * 2;    // swizzled bf16
constexpr size_t OFF_WR    = OFF_WL    + (size_t)INP * UP   * 2;
constexpr size_t OFF_WSK   = OFF_WR    + (size_t)INP * HID  * 2;
constexpr size_t OFF_WF    = OFF_WSK   + (size_t)UP  * HID  * 2;
constexpr size_t OFF_WD    = OFF_WF    + (size_t)UP  * FOUT * 2;
// total ~16.9 MB

// ---------------- grid-wide barrier (sense via generation counter) --------
__device__ __forceinline__ void gsync(int* cnt, int* gen) {
  __syncthreads();
  if (threadIdx.x == 0) {
    __threadfence();
    int g = __hip_atomic_load(gen, __ATOMIC_RELAXED, __HIP_MEMORY_SCOPE_AGENT);
    int arrived = __hip_atomic_fetch_add(cnt, 1, __ATOMIC_ACQ_REL,
                                         __HIP_MEMORY_SCOPE_AGENT) + 1;
    if (arrived == NWG) {
      __hip_atomic_store(cnt, 0, __ATOMIC_RELAXED, __HIP_MEMORY_SCOPE_AGENT);
      __hip_atomic_store(gen, g + 1, __ATOMIC_RELEASE, __HIP_MEMORY_SCOPE_AGENT);
    } else {
      while (__hip_atomic_load(gen, __ATOMIC_ACQUIRE,
                               __HIP_MEMORY_SCOPE_AGENT) == g) {
        __builtin_amdgcn_s_sleep(2);
      }
    }
    __threadfence();
  }
  __syncthreads();
}

// ---------------- WMMA 16x16 N-tile GEMM over K (bf16 in, f32 acc) --------
// A:   u16[16][lda] row-major (rows 8..15 zero).  Lane fragment = two 16B
//      contiguous runs (K {0..7,16..23} or {8..15,24..31} + base).
// Bsw: fragment-swizzled weights: block (ntile, k0/32) is 1KB laid out
//      [lane][16 elems]; lane's 16 bf16 = 32 contiguous bytes.
__device__ __forceinline__ v8f wmma_tile(const unsigned short* __restrict__ A,
                                         int lda,
                                         const unsigned short* __restrict__ Bsw,
                                         int K, int ntile, int lane) {
  v8f acc = {};
  const int m     = lane & 15;
  const int koffA = (lane >> 4) * 8;
  const int KT    = K >> 5;
  const unsigned short* ap = A + m * lda;
  const unsigned short* bp = Bsw + ((size_t)ntile * KT) * 512 + lane * 16;
  for (int k0 = 0; k0 < K; k0 += 32) {
    if (k0 + 64 < K)  // pull next blocks toward the WGP cache
      __builtin_prefetch(bp + 1024, 0, 3);
    FragU a, b;
    a.q[0] = *reinterpret_cast<const uint4*>(ap + k0 + koffA);
    a.q[1] = *reinterpret_cast<const uint4*>(ap + k0 + 16 + koffA);
    b.q[0] = *reinterpret_cast<const uint4*>(bp);
    b.q[1] = *reinterpret_cast<const uint4*>(bp + 8);
    acc = __builtin_amdgcn_wmma_f32_16x16x32_bf16(false, a.v, false, b.v,
                                                  (short)0, acc, false, false);
    bp += 512;
  }
  return acc;
}

// ---------------- LayerNorm (768) -> bf16 row, one workgroup per row ------
__device__ void ln_row_to_bf(const float* __restrict__ row,
                             const float* __restrict__ g,
                             const float* __restrict__ bt,
                             unsigned short* __restrict__ outb,
                             float* red, int tid) {
  float s = 0.f;
  for (int i = tid; i < INP; i += NTH) s += row[i];
  red[tid] = s; __syncthreads();
  for (int o = NTH / 2; o > 0; o >>= 1) {
    if (tid < o) red[tid] += red[tid + o];
    __syncthreads();
  }
  float mu = red[0] * (1.0f / INP);
  __syncthreads();
  float v = 0.f;
  for (int i = tid; i < INP; i += NTH) { float d = row[i] - mu; v += d * d; }
  red[tid] = v; __syncthreads();
  for (int o = NTH / 2; o > 0; o >>= 1) {
    if (tid < o) red[tid] += red[tid + o];
    __syncthreads();
  }
  float rs = rsqrtf(red[0] * (1.0f / INP) + EPSLN);
  __syncthreads();
  for (int i = tid; i < INP; i += NTH)
    outb[i] = f2bf((row[i] - mu) * rs * g[i] + bt[i]);
}

// -------- weight swizzler: src f32 [N][K] row-major (B[k][n]=src[n][k]) ---
// dst block layout: s = ((bn*KT + bk)*32 + lane)*16 + i
//   n = bn*16 + (lane&15);  k = bk*32 + (lane>>4)*8 + i + (i>=8 ? 8 : 0)
__device__ void swz_w(const float* __restrict__ src,
                      unsigned short* __restrict__ dst,
                      int N, int K, size_t gt, size_t st) {
  const size_t total = (size_t)N * K;
  const int KT = K >> 5;
  for (size_t s = gt; s < total; s += st) {
    int i   = (int)(s & 15);
    int ln  = (int)((s >> 4) & 31);
    size_t blk = s >> 9;
    int bk = (int)(blk % KT);
    int bn = (int)(blk / KT);
    int n = bn * 16 + (ln & 15);
    int k = bk * 32 + (ln >> 4) * 8 + i + (i >= 8 ? 8 : 0);
    dst[s] = f2bf(src[(size_t)n * K + k]);
  }
}

// ---------------- init: convert/swizzle weights, reset barrier & state ----
__global__ void mlstm_init(const float* __restrict__ W_up_l,
                           const float* __restrict__ W_up_r,
                           const float* __restrict__ W_skip,
                           const float* __restrict__ fused_W,
                           const float* __restrict__ W_down,
                           char* __restrict__ ws, float* __restrict__ out) {
  size_t gt = (size_t)blockIdx.x * blockDim.x + threadIdx.x;
  size_t st = (size_t)gridDim.x * blockDim.x;
  if (gt == 0) { int* c = (int*)(ws + OFF_SYNC); c[0] = 0; c[1] = 0; }

  swz_w(W_up_l,  (unsigned short*)(ws + OFF_WL),  UP,   INP, gt, st);
  swz_w(W_up_r,  (unsigned short*)(ws + OFF_WR),  HID,  INP, gt, st);
  swz_w(W_skip,  (unsigned short*)(ws + OFF_WSK), HID,  UP,  gt, st);
  swz_w(fused_W, (unsigned short*)(ws + OFF_WF),  FOUT, UP,  gt, st);
  swz_w(W_down,  (unsigned short*)(ws + OFF_WD),  INP,  HID, gt, st);

  // zero the bf16 A-staging buffers (pad rows 8..15 must be zero)
  unsigned short* xnb = (unsigned short*)(ws + OFF_XNB);
  unsigned short* xcb = (unsigned short*)(ws + OFF_XCB);
  unsigned short* hnb = (unsigned short*)(ws + OFF_HNB);
  for (size_t i = gt; i < (size_t)16 * INP; i += st) xnb[i] = 0;
  for (size_t i = gt; i < (size_t)16 * UP;  i += st) xcb[i] = 0;
  for (size_t i = gt; i < (size_t)16 * HID; i += st) hnb[i] = 0;

  // state init in d_out: C=0, n=1, m=0
  float* Cst = out + (size_t)B_ * S_ * INP;
  float* nst = Cst + (size_t)B_ * H_ * D_ * D_;
  float* mst = nst + (size_t)B_ * H_ * D_;
  for (size_t i = gt; i < (size_t)B_ * H_ * D_ * D_; i += st) Cst[i] = 0.f;
  for (size_t i = gt; i < (size_t)B_ * H_ * D_;      i += st) nst[i] = 1.f;
  for (size_t i = gt; i < (size_t)B_ * H_;           i += st) mst[i] = 0.f;
}

// ---------------- persistent scan kernel ----------------------------------
__global__ __launch_bounds__(NTH)
void mlstm_persistent(const float* __restrict__ x,
                      const float* __restrict__ ln1_g,
                      const float* __restrict__ ln1_b,
                      const float* __restrict__ conv_w,
                      const float* __restrict__ fused_bias,
                      const float* __restrict__ ln2_g,
                      const float* __restrict__ ln2_b,
                      char* __restrict__ ws, float* __restrict__ out) {
  const int tid  = threadIdx.x;
  const int wg   = blockIdx.x;
  const int lane = tid & 31;
  const int gw   = wg * (NTH / 32) + (tid >> 5);   // global wave id, 0..511

  int* cnt = (int*)(ws + OFF_SYNC);
  int* gen = cnt + 1;
  unsigned short* xnb  = (unsigned short*)(ws + OFF_XNB);
  float*          xt   = (float*)(ws + OFF_XT);
  float*          rt   = (float*)(ws + OFF_RT);
  unsigned short* xcb  = (unsigned short*)(ws + OFF_XCB);
  float*          fusd = (float*)(ws + OFF_FUSED);
  float*          xsk  = (float*)(ws + OFF_XSKIP);
  float*          hbuf = (float*)(ws + OFF_HBUF);
  unsigned short* hnb  = (unsigned short*)(ws + OFF_HNB);
  const unsigned short* Wl  = (const unsigned short*)(ws + OFF_WL);
  const unsigned short* Wr  = (const unsigned short*)(ws + OFF_WR);
  const unsigned short* Wsk = (const unsigned short*)(ws + OFF_WSK);
  const unsigned short* Wf  = (const unsigned short*)(ws + OFF_WF);
  const unsigned short* Wd  = (const unsigned short*)(ws + OFF_WD);

  float* ys  = out;
  float* Cst = out + (size_t)B_ * S_ * INP;
  float* nst = Cst + (size_t)B_ * H_ * D_ * D_;
  float* mst = nst + (size_t)B_ * H_ * D_;

  __shared__ float smem[640];

  const float w0 = conv_w[0], w1 = conv_w[1], w2 = conv_w[2], w3 = conv_w[3];

  for (int t = 0; t < S_; ++t) {
    // ---- P0: LN1 of x[:,t,:] -> bf16 A-matrix (rows 0..7) ----
    if (wg < B_) {
      const float* row = x + ((size_t)wg * S_ + t) * INP;
      ln_row_to_bf(row, ln1_g, ln1_b, xnb + wg * INP, smem, tid);
    }
    gsync(cnt, gen);

    // ---- P1: xt = x_n @ W_up_l^T ; rt = x_n @ W_up_r^T ----
    if (gw < UP / 16) {
      int n0 = gw * 16;
      v8f acc = wmma_tile(xnb, INP, Wl, INP, gw, lane);
      if (lane < 16) {
#pragma unroll
        for (int r = 0; r < 8; ++r) xt[r * UP + n0 + lane] = acc[r];
      }
    } else if (gw < UP / 16 + HID / 16) {
      int ntile = gw - UP / 16;
      int n0 = ntile * 16;
      v8f acc = wmma_tile(xnb, INP, Wr, INP, ntile, lane);
      if (lane < 16) {
#pragma unroll
        for (int r = 0; r < 8; ++r) rt[r * HID + n0 + lane] = acc[r];
      }
    }
    gsync(cnt, gen);

    // ---- P2: causal 4-tap conv over features + SiLU -> bf16 ----
    for (int e = wg * NTH + tid; e < B_ * UP; e += NWG * NTH) {
      int b = e / UP, u = e % UP;
      const float* xr = xt + b * UP;
      float s = w3 * xr[u];
      if (u >= 1) s += w2 * xr[u - 1];
      if (u >= 2) s += w1 * xr[u - 2];
      if (u >= 3) s += w0 * xr[u - 3];
      float sg = 1.f / (1.f + __expf(-s));
      xcb[b * UP + u] = f2bf(s * sg);
    }
    gsync(cnt, gen);

    // ---- P3: fused = xc @ fused_W^T + b ; x_skip = xc @ W_skip^T ----
    if (gw < FOUT / 16) {
      int n0 = gw * 16;
      v8f acc = wmma_tile(xcb, UP, Wf, UP, gw, lane);
      if (lane < 16) {
        float bb = fused_bias[n0 + lane];
#pragma unroll
        for (int r = 0; r < 8; ++r) fusd[r * FOUT + n0 + lane] = acc[r] + bb;
      }
    } else if (gw < FOUT / 16 + HID / 16) {
      int ntile = gw - FOUT / 16;
      int n0 = ntile * 16;
      v8f acc = wmma_tile(xcb, UP, Wsk, UP, ntile, lane);
      if (lane < 16) {
#pragma unroll
        for (int r = 0; r < 8; ++r) xsk[r * HID + n0 + lane] = acc[r];
      }
    }
    gsync(cnt, gen);

    // ---- P4: gates + mLSTM state update; one (b,h) head per workgroup ----
    {
      const int b = wg >> 3, h = wg & 7;
      const float* fb = fusd + b * FOUT;
      float m_prev = mst[b * H_ + h];
      float i_t = CAPG * tanhf(fb[h] * (1.0f / CAPG));
      float f_t = CAPG * tanhf(fb[H_ + h] * (1.0f / CAPG));
      float m_t = fmaxf(f_t + m_prev, i_t);
      float ie = __expf(i_t - m_t);
      float fe = __expf(f_t - m_t + m_prev);

      float* qq = smem;           // 96
      float* kk = smem + 96;      // 96
      float* vv = smem + 192;     // 96
      float* hp = smem + 288;     // 192 (row partials for C@q)
      float* dr = smem + 480;     // 96  (n.q partials)
      const float invsq = 0.10206207261596575f;  // 1/sqrt(96)

      if (tid < D_) {
        qq[tid] = fb[2 * H_ + HID + h * D_ + tid];
        kk[tid] = fb[2 * H_ + 2 * HID + h * D_ + tid] * invsq;
        vv[tid] = fb[2 * H_ + 3 * HID + h * D_ + tid];
      }
      __syncthreads();

      float* Cw = Cst + (size_t)(b * H_ + h) * D_ * D_;
      if (tid < 2 * D_) {                 // 2 threads per C row
        int d = tid >> 1, half = tid & 1;
        float vd = vv[d] * ie;
        float acc = 0.f;
        int e0 = half * (D_ / 2);
        for (int e = e0; e < e0 + D_ / 2; ++e) {
          float c = fe * Cw[d * D_ + e] + vd * kk[e];
          Cw[d * D_ + e] = c;
          acc += c * qq[e];
        }
        hp[tid] = acc;
      }
      if (tid < D_) {
        float* nw = nst + (b * H_ + h) * D_;
        float nn = fe * nw[tid] + ie * kk[tid];
        nw[tid] = nn;
        dr[tid] = nn * qq[tid];
      }
      __syncthreads();
      if (tid == 0) {
        float s = 0.f;
        for (int i = 0; i < D_; ++i) s += dr[i];
        smem[576] = fmaxf(s, 1.0f);
        mst[b * H_ + h] = m_t;
      }
      __syncthreads();
      if (tid < D_) {
        float den = smem[576];
        float oc = CAPG * tanhf(fb[2 * H_ + h * D_ + tid] * (1.0f / CAPG));
        float o  = 1.f / (1.f + __expf(-oc));
        float hn = hp[2 * tid] + hp[2 * tid + 1];
        hbuf[b * HID + h * D_ + tid] = o * hn / den;
      }
    }
    gsync(cnt, gen);

    // ---- P5: LN2(h) -> bf16 ----
    if (wg < B_) ln_row_to_bf(hbuf + wg * HID, ln2_g, ln2_b, hnb + wg * HID,
                              smem, tid);
    gsync(cnt, gen);

    // ---- P6: down GEMM + fused epilogue -> ys[:,t,:] ----
    if (gw < INP / 16) {
      int n0 = gw * 16;
      v8f acc = wmma_tile(hnb, HID, Wd, HID, gw, lane);
      if (lane < 16) {
        int n = n0 + lane;
#pragma unroll
        for (int r = 0; r < 8; ++r) {
          float v = acc[r] + xsk[r * HID + n];
          float rr = rt[r * HID + n];
          v *= rr / (1.f + __expf(-rr));          // * silu(rt)
          size_t xi = ((size_t)r * S_ + t) * INP + n;
          ys[xi] = v + x[xi];                     // + residual
        }
      }
    }
    gsync(cnt, gen);
  }
}

// ---------------------------------------------------------------------------
extern "C" void kernel_launch(void* const* d_in, const int* in_sizes, int n_in,
                              void* d_out, int out_size, void* d_ws,
                              size_t ws_size, hipStream_t stream) {
  const float* x       = (const float*)d_in[0];
  const float* ln1_g   = (const float*)d_in[1];
  const float* ln1_b   = (const float*)d_in[2];
  const float* W_up_l  = (const float*)d_in[3];
  const float* W_up_r  = (const float*)d_in[4];
  const float* conv_w  = (const float*)d_in[5];
  const float* W_skip  = (const float*)d_in[6];
  const float* fused_W = (const float*)d_in[7];
  const float* fused_b = (const float*)d_in[8];
  const float* ln2_g   = (const float*)d_in[9];
  const float* ln2_b   = (const float*)d_in[10];
  const float* W_down  = (const float*)d_in[11];
  char*  ws  = (char*)d_ws;
  float* out = (float*)d_out;

  hipLaunchKernelGGL(mlstm_init, dim3(1024), dim3(256), 0, stream,
                     W_up_l, W_up_r, W_skip, fused_W, W_down, ws, out);
  hipLaunchKernelGGL(mlstm_persistent, dim3(NWG), dim3(NTH), 0, stream,
                     x, ln1_g, ln1_b, conv_w, fused_b, ln2_g, ln2_b, ws, out);
}
